// GroupedSidechannelCrossAttention_74543452389416
// MI455X (gfx1250) — matrix operand order repaired and checked by reading, not compiled, not run on hardware
//
#include <hip/hip_runtime.h>
#include <hip/hip_bf16.h>
#include <math.h>

// ---------------------------------------------------------------------------
// Types for CDNA5 WMMA (wave32)
// ---------------------------------------------------------------------------
typedef __bf16 bf16_t;
typedef __attribute__((ext_vector_type(16))) __bf16 v16bf;
typedef __attribute__((ext_vector_type(8)))  __bf16 v8bf;
typedef __attribute__((ext_vector_type(8)))  float  v8f;
typedef int v4i __attribute__((vector_size(16)));

// Problem constants (match reference)
#define BB 2
#define TT 1024
#define CC 2048
#define NCH 8
#define KLAT 32
#define NHEAD 16
#define NKV 4
#define DD 128
#define EKV (NKV * DD)      // 512
#define BT (BB * TT)        // 2048

// ---------------------------------------------------------------------------
// gfx1250 async global->LDS copy (ASYNCcnt-tracked), 16B per lane.
// Builtin signature (probe-confirmed this round):
//   (v4i addrspace(1)* src, v4i addrspace(3)* dst, imm offset, imm cpol)
// ---------------------------------------------------------------------------
#define GP4(p) ((__attribute__((address_space(1))) v4i*)(p))
#define LP4(p) ((__attribute__((address_space(3))) v4i*)(p))

__device__ __forceinline__ void async_cp16(void* lds_dst, const void* gsrc) {
#if __has_builtin(__builtin_amdgcn_global_load_async_to_lds_b128)
  __builtin_amdgcn_global_load_async_to_lds_b128(GP4((void*)gsrc), LP4(lds_dst), 0, 0);
#else
  unsigned lds_off = (unsigned)(__UINTPTR_TYPE__)LP4(lds_dst);
  unsigned long long ga = (unsigned long long)(__UINTPTR_TYPE__)gsrc;
  asm volatile("global_load_async_to_lds_b128 %0, %1, off"
               :: "v"(lds_off), "v"(ga) : "memory");
#endif
}

__device__ __forceinline__ void wait_async0() {
#if __has_builtin(__builtin_amdgcn_s_wait_asynccnt)
  __builtin_amdgcn_s_wait_asynccnt(0);
#else
  asm volatile("s_wait_asynccnt 0x0" ::: "memory");
#endif
}

// ---------------------------------------------------------------------------
// fp32 -> bf16 elementwise convert (vector-of-4 per thread)
// ---------------------------------------------------------------------------
__global__ void cvt_f32_bf16_kernel(const float* __restrict__ src,
                                    bf16_t* __restrict__ dst, int n) {
  int i4 = (blockIdx.x * 256 + threadIdx.x) * 4;
  if (i4 + 3 < n) {
    float4 f = *(const float4*)&src[i4];
    bf16_t o[4];
    o[0] = (bf16_t)f.x; o[1] = (bf16_t)f.y; o[2] = (bf16_t)f.z; o[3] = (bf16_t)f.w;
    *(uint2*)&dst[i4] = *(const uint2*)o;
  }
}

// ---------------------------------------------------------------------------
// fp32 (R x E, row-major) -> bf16 transposed (E x R, row-major)
// ---------------------------------------------------------------------------
__global__ void transpose_cvt_kernel(const float* __restrict__ W,
                                     bf16_t* __restrict__ WT, int R, int E) {
  __shared__ float tile[32][33];
  int c0 = blockIdx.x * 32;
  int r0 = blockIdx.y * 32;
  int tx = threadIdx.x, ty = threadIdx.y;
  #pragma unroll
  for (int i = 0; i < 32; i += 8)
    tile[ty + i][tx] = W[(size_t)(r0 + ty + i) * E + (c0 + tx)];
  __syncthreads();
  #pragma unroll
  for (int i = 0; i < 32; i += 8)
    WT[(size_t)(c0 + ty + i) * R + (r0 + tx)] = (bf16_t)tile[tx][ty + i];
}

// ---------------------------------------------------------------------------
// WMMA bf16 GEMM:  C[M,N] (fp32) = A[M,K] (bf16, row-major) x Bt[N,K]^T
// Block tile 128x128, BK=32, 128 threads = 4 waves (wave32), 2x2 wave grid,
// each wave computes 64x64 = 4x4 WMMA subtiles.
// Double-buffered LDS, async global->LDS staging, one barrier per k-step.
// EPI==1: multiply output by tanhf(gate[0]).
// ---------------------------------------------------------------------------
#define LDS_STRIDE 40   // 32 + 8 bf16 pad (row start = 80 B, 16B-aligned)
#define TILE_ELEMS (128 * LDS_STRIDE)

__device__ __forceinline__ v16bf ld_frag(const bf16_t* p0, const bf16_t* p1) {
  union { v16bf v; v8bf h[2]; } u;
  u.h[0] = *(const v8bf*)p0;
  u.h[1] = *(const v8bf*)p1;
  return u.v;
}

// stage one 128x32 A-slab and 128x32 Bt-slab into LDS buffers (async)
__device__ __forceinline__ void stage_tiles(const bf16_t* __restrict__ A,
                                            const bf16_t* __restrict__ Bt,
                                            bf16_t* As, bf16_t* Bs,
                                            int tileM, int tileN, int K, int k0,
                                            int tid) {
  #pragma unroll
  for (int i = 0; i < 4; ++i) {
    int c   = tid + 128 * i;     // 0..511
    int row = c >> 2;            // 0..127
    int off = (c & 3) * 8;       // 0,8,16,24 bf16 (16B chunks)
    async_cp16(&As[row * LDS_STRIDE + off],
               &A[(size_t)(tileM + row) * K + k0 + off]);
    async_cp16(&Bs[row * LDS_STRIDE + off],
               &Bt[(size_t)(tileN + row) * K + k0 + off]);
  }
}

__device__ __forceinline__ void compute_step(const bf16_t* As, const bf16_t* Bs,
                                             v8f acc[4][4],
                                             int wm, int wn, int lhalf,
                                             int koffA, int koffB) {
  v16bf bfrag[4];
  #pragma unroll
  for (int j = 0; j < 4; ++j) {
    int nr = wn + j * 16 + lhalf;
    bfrag[j] = ld_frag(&Bs[nr * LDS_STRIDE + koffB],
                       &Bs[nr * LDS_STRIDE + koffB + 16]);
  }
  #pragma unroll
  for (int i = 0; i < 4; ++i) {
    int rb = wm + i * 16 + lhalf;
    v16bf afrag = ld_frag(&As[rb * LDS_STRIDE + koffA],
                          &As[rb * LDS_STRIDE + koffA + 16]);
    #pragma unroll
    for (int j = 0; j < 4; ++j) {
      acc[i][j] = __builtin_amdgcn_wmma_f32_16x16x32_bf16(
          false, afrag, false, bfrag[j], (short)0, acc[i][j], false, false);
    }
  }
}

template <int EPI>
__global__ __launch_bounds__(128)
void gemm_bf16_wmma_kernel(const bf16_t* __restrict__ A,
                           const bf16_t* __restrict__ Bt,
                           float* __restrict__ C,
                           int M, int N, int K,
                           const float* __restrict__ gate) {
  __shared__ bf16_t As0[TILE_ELEMS], Bs0[TILE_ELEMS];
  __shared__ bf16_t As1[TILE_ELEMS], Bs1[TILE_ELEMS];

  const int tid   = threadIdx.x;
  const int tileM = blockIdx.y * 128;
  const int tileN = blockIdx.x * 128;

  const int wid   = tid >> 5;
  const int lane  = tid & 31;
  const int wm    = (wid & 1) * 64;   // wave m-offset inside block tile
  const int wn    = (wid >> 1) * 64;  // wave n-offset inside block tile
  const int lhalf = lane & 15;
  const bool hi   = lane >= 16;
  const int koffA = hi ? 8 : 0;       // A frag K sub-offset per ISA layout
  const int koffB = hi ? 8 : 0;       // B frag K sub-offset per ISA layout

  v8f acc[4][4] = {};

  // prologue: stage k=0 into buffer 0
  stage_tiles(A, Bt, As0, Bs0, tileM, tileN, K, 0, tid);

  // K is a multiple of 64: 2x-unrolled double-buffered pipeline
  for (int k0 = 0; k0 < K; k0 += 64) {
    wait_async0();                 // my async writes into buf0 are done
    __syncthreads();               // everyone's buf0 ready; buf1 free
    stage_tiles(A, Bt, As1, Bs1, tileM, tileN, K, k0 + 32, tid);
    compute_step(As0, Bs0, acc, wm, wn, lhalf, koffA, koffB);

    wait_async0();                 // buf1 ready
    __syncthreads();               // all waves done reading buf0
    if (k0 + 64 < K)
      stage_tiles(A, Bt, As0, Bs0, tileM, tileN, K, k0 + 64, tid);
    compute_step(As1, Bs1, acc, wm, wn, lhalf, koffA, koffB);
  }

  float scale = 1.0f;
  if (EPI == 1) scale = tanhf(gate[0]);

  // C/D layout: VGPR r, lane<16 -> row r, col lane; lane>=16 -> row r+8
  #pragma unroll
  for (int i = 0; i < 4; ++i) {
    #pragma unroll
    for (int j = 0; j < 4; ++j) {
      #pragma unroll
      for (int r = 0; r < 8; ++r) {
        int row = tileM + wm + i * 16 + r + (hi ? 8 : 0);
        int col = tileN + wn + j * 16 + lhalf;
        C[(size_t)row * N + col] = acc[i][j][r] * scale;
      }
    }
  }
}

// ---------------------------------------------------------------------------
// Tiny-K attention: one block per token (256 thr = 8 waves, 2 heads/wave).
// Qf fp32 (BT x 2048); Kp/Vp fp32 (B*N*K x 512); writes yb bf16 (BT x 2048).
// ---------------------------------------------------------------------------
__global__ __launch_bounds__(256)
void attn_kernel(const float* __restrict__ Qf,
                 const float* __restrict__ Kp,
                 const float* __restrict__ Vp,
                 const int* __restrict__ mask,
                 bf16_t* __restrict__ yb) {
  __shared__ float qs[CC];
  __shared__ int nsh;

  const int bt = blockIdx.x;       // 0 .. B*T-1
  const int b  = bt >> 10;         // T = 1024
  const int tid = threadIdx.x;

  for (int c = tid; c < CC / 4; c += 256)
    *(float4*)&qs[c * 4] = *(const float4*)&Qf[(size_t)bt * CC + c * 4];
  if (tid == 0) nsh = mask[bt];
  __syncthreads();

  const int n = nsh;
  const int wid = tid >> 5, lane = tid & 31;
  const float scale = 0.08838834764831845f;  // 1/sqrt(128)
  const size_t kvbase = (size_t)(b * NCH + n) * KLAT * EKV;

  #pragma unroll
  for (int hh = 0; hh < 2; ++hh) {
    const int h = wid * 2 + hh;
    const int g = h >> 2;
    // scores: lane == latent j, dot over D=128
    const float* krow = Kp + kvbase + (size_t)lane * EKV + g * DD;
    float s = 0.f;
    #pragma unroll 8
    for (int d = 0; d < DD; d += 4) {
      float4 k4 = *(const float4*)&krow[d];
      float4 q4 = *(const float4*)&qs[h * DD + d];
      s = fmaf(q4.x, k4.x, s); s = fmaf(q4.y, k4.y, s);
      s = fmaf(q4.z, k4.z, s); s = fmaf(q4.w, k4.w, s);
    }
    s *= scale;
    // wave32 softmax over 32 latents
    float m = s;
    #pragma unroll
    for (int o = 16; o > 0; o >>= 1) m = fmaxf(m, __shfl_xor(m, o, 32));
    float e = __expf(s - m);
    float sum = e;
    #pragma unroll
    for (int o = 16; o > 0; o >>= 1) sum += __shfl_xor(sum, o, 32);
    float a = e / sum;
    // output: lane owns d = lane + 32*u
    float y0 = 0.f, y1 = 0.f, y2 = 0.f, y3 = 0.f;
    const float* vbase = Vp + kvbase + g * DD;
    #pragma unroll 4
    for (int j = 0; j < KLAT; ++j) {
      float aj = __shfl(a, j, 32);
      const float* vrow = vbase + (size_t)j * EKV;
      y0 = fmaf(aj, vrow[lane], y0);
      y1 = fmaf(aj, vrow[lane + 32], y1);
      y2 = fmaf(aj, vrow[lane + 64], y2);
      y3 = fmaf(aj, vrow[lane + 96], y3);
    }
    size_t ob = (size_t)bt * CC + h * DD;
    yb[ob + lane]      = (bf16_t)y0;
    yb[ob + lane + 32] = (bf16_t)y1;
    yb[ob + lane + 64] = (bf16_t)y2;
    yb[ob + lane + 96] = (bf16_t)y3;
  }
}

// ---------------------------------------------------------------------------
// Launcher
// Inputs: 0:x 1:channel_states 2:channel_mask 3:Wq 4:Wk 5:Wv 6:Wo 7:gate
// ---------------------------------------------------------------------------
extern "C" void kernel_launch(void* const* d_in, const int* in_sizes, int n_in,
                              void* d_out, int out_size, void* d_ws, size_t ws_size,
                              hipStream_t stream) {
  const float* x   = (const float*)d_in[0];
  const float* cs  = (const float*)d_in[1];
  const int*   cm  = (const int*)d_in[2];
  const float* Wq  = (const float*)d_in[3];
  const float* Wk  = (const float*)d_in[4];
  const float* Wv  = (const float*)d_in[5];
  const float* Wo  = (const float*)d_in[6];
  const float* gate = (const float*)d_in[7];
  float* out = (float*)d_out;

  // workspace carve-up
  char* ws = (char*)d_ws;
  size_t off = 0;
  bf16_t* xb  = (bf16_t*)(ws + off); off += (size_t)BT * CC * 2;        // 8 MB
  bf16_t* csb = (bf16_t*)(ws + off); off += (size_t)BB * NCH * KLAT * CC * 2; // 2 MB
  bf16_t* WqT = (bf16_t*)(ws + off); off += (size_t)CC * CC * 2;        // 8 MB
  bf16_t* WkT = (bf16_t*)(ws + off); off += (size_t)EKV * CC * 2;       // 2 MB
  bf16_t* WvT = (bf16_t*)(ws + off); off += (size_t)EKV * CC * 2;       // 2 MB
  bf16_t* WoT = (bf16_t*)(ws + off); off += (size_t)CC * CC * 2;        // 8 MB
  float*  Qf  = (float*)(ws + off);  off += (size_t)BT * CC * 4;        // 16 MB
  float*  Kp  = (float*)(ws + off);  off += (size_t)BB * NCH * KLAT * EKV * 4; // 1 MB
  float*  Vp  = (float*)(ws + off);  off += (size_t)BB * NCH * KLAT * EKV * 4; // 1 MB
  bf16_t* yb  = (bf16_t*)(ws + off); off += (size_t)BT * CC * 2;        // 8 MB

  // 1) precision conversion of activations / latents
  {
    int n = BT * CC;                       // 4,194,304
    cvt_f32_bf16_kernel<<<n / (256 * 4), 256, 0, stream>>>(x, xb, n);
    n = BB * NCH * KLAT * CC;              // 1,048,576
    cvt_f32_bf16_kernel<<<n / (256 * 4), 256, 0, stream>>>(cs, csb, n);
  }
  // 2) weight convert+transpose (W is C x E row-major -> WT is E x C)
  {
    dim3 blk(32, 8);
    transpose_cvt_kernel<<<dim3(CC / 32,  CC / 32), blk, 0, stream>>>(Wq, WqT, CC, CC);
    transpose_cvt_kernel<<<dim3(EKV / 32, CC / 32), blk, 0, stream>>>(Wk, WkT, CC, EKV);
    transpose_cvt_kernel<<<dim3(EKV / 32, CC / 32), blk, 0, stream>>>(Wv, WvT, CC, EKV);
    transpose_cvt_kernel<<<dim3(CC / 32,  CC / 32), blk, 0, stream>>>(Wo, WoT, CC, CC);
  }
  // 3) Q projection: (2048 x 2048) = xb @ Wq
  gemm_bf16_wmma_kernel<0><<<dim3(CC / 128, BT / 128), 128, 0, stream>>>(
      xb, WqT, Qf, BT, CC, CC, nullptr);
  // 4) latent K/V projections: (512 x 512) = csb @ Wk / Wv
  {
    const int MR = BB * NCH * KLAT;  // 512
    gemm_bf16_wmma_kernel<0><<<dim3(EKV / 128, MR / 128), 128, 0, stream>>>(
        csb, WkT, Kp, MR, EKV, CC, nullptr);
    gemm_bf16_wmma_kernel<0><<<dim3(EKV / 128, MR / 128), 128, 0, stream>>>(
        csb, WvT, Vp, MR, EKV, CC, nullptr);
  }
  // 5) per-token tiny-K attention -> yb (bf16)
  attn_kernel<<<BT, 256, 0, stream>>>(Qf, Kp, Vp, cm, yb);
  // 6) output projection with tanh(gate) epilogue
  gemm_bf16_wmma_kernel<1><<<dim3(CC / 128, BT / 128), 128, 0, stream>>>(
      yb, WoT, out, BT, CC, CC, gate);
}
